// MinGRUCell_86603720556886
// MI455X (gfx1250) — matrix operand order repaired and checked by reading, not compiled, not run on hardware
//
#include <hip/hip_runtime.h>
#include <hip/hip_bf16.h>
#include <math.h>

typedef __attribute__((ext_vector_type(16))) __bf16 v16bf;
typedef __attribute__((ext_vector_type(8)))  __bf16 v8bf;
typedef __attribute__((ext_vector_type(8)))  float  v8f;
typedef __attribute__((ext_vector_type(4)))  float  v4f;

#define B_    8
#define S_    4096
#define D_    1024
#define H_    1024
#define MTOT  (B_ * S_)      // 32768 rows of x (flattened B,S)
#define BH    (B_ * H_)      // 8192 independent scan channels
#define WN    (H_ * D_)      // elements per weight matrix
#define CHUNKS 32
#define CS     (S_ / CHUNKS) // 128 steps per chunk

// GEMM tiling: block = 256 threads (8 waves) = (mat, msub)
// wave: 2 M-tiles (rows msub*16 and +64) x 128 cols of one matrix
#define BLK_M 128
#define BLK_N 128
#define NT_W  8              // 16-col N-tiles per wave

// Fast, stable log-space helpers (hardware v_exp_f32 / v_log_f32)
__device__ __forceinline__ float log1pexp_neg(float t) {   // t <= 0
    return __logf(1.0f + __expf(t));
}
__device__ __forceinline__ float softplusf(float v) {
    return fmaxf(v, 0.0f) + log1pexp_neg(-fabsf(v));
}
__device__ __forceinline__ float laef(float a, float b) {  // logaddexp
    return fmaxf(a, b) + log1pexp_neg(-fabsf(a - b));
}
__device__ __forceinline__ float log_g(float x) {
    return (x >= 0.0f) ? __logf(x + 0.5f) : -softplusf(-x);
}
// k,g -> log_coeffs, log_values (one exp+log serves softplus(k) AND softplus(-k))
__device__ __forceinline__ void kg_transform(float k, float g, float& lcv, float& lvv) {
    const float u  = __logf(1.0f + __expf(-fabsf(k)));
    lcv = -(fmaxf(k, 0.0f) + u);                  // -softplus(k)
    const float lz = -(fmaxf(-k, 0.0f) + u);      // -softplus(-k)
    lvv = lz + log_g(g);
}

// ---------------------------------------------------------------------------
// Kernel 0a: x fp32 -> bf16 (row-major, same layout)
// ---------------------------------------------------------------------------
__global__ __launch_bounds__(256)
void convert_x_kernel(const float* __restrict__ x, __bf16* __restrict__ xbf)
{
    const size_t i = ((size_t)blockIdx.x * 256 + threadIdx.x) * 8;   // < MTOT*D_
    v4f a = *(const v4f*)(x + i);
    v4f b = *(const v4f*)(x + i + 4);
    v8bf o;
    o[0] = (__bf16)a[0]; o[1] = (__bf16)a[1]; o[2] = (__bf16)a[2]; o[3] = (__bf16)a[3];
    o[4] = (__bf16)b[0]; o[5] = (__bf16)b[1]; o[6] = (__bf16)b[2]; o[7] = (__bf16)b[3];
    *(v8bf*)(xbf + i) = o;
}

// ---------------------------------------------------------------------------
// Kernel 0b: W fp32 -> bf16, swizzled to WMMA B-fragment-major layout:
//   frag f = ((mat*64 + ntg)*32 + ks), addr = f*512 + lane*16  (bf16 elems)
//   lane l: row = ntg*16 + (l&15); elems 0-15 = k0+(l>>4)*16+j
//   (same K-permutation as the A fragment -> consistent WMMA reduction)
// ---------------------------------------------------------------------------
__global__ __launch_bounds__(256)
void swizzle_w_kernel(const float* __restrict__ Wz, const float* __restrict__ Wh,
                      __bf16* __restrict__ wswz)
{
    const int gid  = blockIdx.x * 256 + threadIdx.x;   // < 2*64*32*32 = 131072
    const int lane = gid & 31;
    const int ks   = (gid >> 5) & 31;
    const int ntg  = (gid >> 10) & 63;
    const int mat  = gid >> 16;
    const float* W = mat ? Wh : Wz;
    const int row  = ntg * 16 + (lane & 15);
    const int k0   = ks * 32 + (lane >> 4) * 16;
    const float* src = W + (size_t)row * D_;

    v4f a0 = *(const v4f*)(src + k0);
    v4f a1 = *(const v4f*)(src + k0 + 4);
    v4f a2 = *(const v4f*)(src + k0 + 8);
    v4f a3 = *(const v4f*)(src + k0 + 12);
    v16bf o;
    #pragma unroll
    for (int j = 0; j < 4; ++j) {
        o[j]      = (__bf16)a0[j];
        o[4 + j]  = (__bf16)a1[j];
        o[8 + j]  = (__bf16)a2[j];
        o[12 + j] = (__bf16)a3[j];
    }
    *(v16bf*)(wswz + (size_t)gid * 16) = o;
}

// ---------------------------------------------------------------------------
// Kernel 1: barrier-free dual GEMM, bf16 WMMA fp32-accumulate.
//   wave = (mat = w&1, msub = w>>1); 2 M-tiles per wave, 8 N-tiles, 16 WMMA/Kstep
//   writes raw k -> karr (mat 0) or g -> garr (mat 1), bias added, NT stores
// ---------------------------------------------------------------------------
__global__ __launch_bounds__(256)
void mingru_gemm_kernel(const __bf16* __restrict__ xbf,
                        const __bf16* __restrict__ wswz,
                        const float* __restrict__ bz, const float* __restrict__ bh,
                        float* __restrict__ karr, float* __restrict__ garr)
{
    const int lane  = threadIdx.x & 31;
    const int wave  = threadIdx.x >> 5;
    const int mat   = wave & 1;
    const int msub  = wave >> 1;
    const int half  = lane >> 4;
    const int lan15 = lane & 15;

    const int m_base    = blockIdx.x * BLK_M + msub * 16;   // second tile at +64
    const int ncol_base = blockIdx.y * BLK_N;
    const int ntg_base  = blockIdx.y * NT_W;

    v8f acc0[NT_W] = {};
    v8f acc1[NT_W] = {};

    // A: lane -> row (lane&15), 16 contiguous bf16 at k = ks*32 + half*16
    const __bf16* xrow0 = xbf + (size_t)(m_base + lan15) * D_ + half * 16;
    const __bf16* xrow1 = xrow0 + (size_t)64 * D_;
    // B: fragment-major, one v16bf (32B) per lane per fragment
    const __bf16* wbase = wswz + (((size_t)mat * 64 + ntg_base) * 32) * 512 + lane * 16;

    #pragma unroll 2
    for (int ks = 0; ks < 32; ++ks) {
        // ---- load phase: 2 A frags + all 8 B frags into live registers
        const v16bf a0 = *(const v16bf*)(xrow0 + ks * 32);
        const v16bf a1 = *(const v16bf*)(xrow1 + ks * 32);
        v16bf bfr[NT_W];
        #pragma unroll
        for (int nt = 0; nt < NT_W; ++nt)
            bfr[nt] = *(const v16bf*)(wbase + ((size_t)nt * 32 + ks) * 512);
        // ---- compute phase: each B fragment feeds both M-tiles
        #pragma unroll
        for (int nt = 0; nt < NT_W; ++nt) {
            acc0[nt] = __builtin_amdgcn_wmma_f32_16x16x32_bf16(
                false, a0, false, bfr[nt], (short)0, acc0[nt], false, false);
            acc1[nt] = __builtin_amdgcn_wmma_f32_16x16x32_bf16(
                false, a1, false, bfr[nt], (short)0, acc1[nt], false, false);
        }
    }

    // Epilogue: C/D layout -> lane l, elem r : M = r + 8*(l>=16), N = l&15
    const float* bias = mat ? bh : bz;
    float* outp       = mat ? garr : karr;
    #pragma unroll
    for (int tile = 0; tile < 2; ++tile) {
        const int crow = m_base + tile * 64 + half * 8;
        #pragma unroll
        for (int nt = 0; nt < NT_W; ++nt) {
            const int col = ncol_base + nt * 16 + lan15;
            const float bv = bias[col];
            const v8f* accp = tile ? &acc1[nt] : &acc0[nt];
            #pragma unroll
            for (int r = 0; r < 8; ++r) {
                const float v = (*accp)[r] + bv;
                __builtin_nontemporal_store(v, &outp[(size_t)(crow + r) * H_ + col]);
            }
        }
    }
}

// ---------------------------------------------------------------------------
// Chunked log-space scan (3 passes), log transforms fused on load
// ---------------------------------------------------------------------------
__global__ __launch_bounds__(256)
void scan_pass1(const float* __restrict__ karr, const float* __restrict__ garr,
                float* __restrict__ Asum, float* __restrict__ Lend)
{
    const int gid = blockIdx.x * blockDim.x + threadIdx.x;   // < BH*CHUNKS
    const int c  = gid / BH;
    const int bh = gid % BH;
    const int b  = bh / H_;
    const int h  = bh % H_;
    size_t e = (size_t)b * S_ * H_ + (size_t)(c * CS) * H_ + h;

    float A = 0.0f, L = -INFINITY;
    #pragma unroll 4
    for (int s = 0; s < CS; ++s, e += H_) {
        const float kk = __builtin_nontemporal_load(&karr[e]);
        const float gg = __builtin_nontemporal_load(&garr[e]);
        float lcv, lvv;
        kg_transform(kk, gg, lcv, lvv);
        A += lcv;
        L = laef(lcv + L, lvv);
    }
    Asum[gid] = A;
    Lend[gid] = L;
}

__global__ __launch_bounds__(256)
void scan_pass2(const float* __restrict__ h0,
                const float* __restrict__ Asum, const float* __restrict__ Lend,
                float* __restrict__ carry)
{
    const int bh = blockIdx.x * blockDim.x + threadIdx.x;    // < BH
    float L = log_g(h0[bh]);
    #pragma unroll
    for (int c = 0; c < CHUNKS; ++c) {
        carry[c * BH + bh] = L;
        L = laef(Asum[c * BH + bh] + L, Lend[c * BH + bh]);
    }
}

__global__ __launch_bounds__(256)
void scan_pass3(const float* __restrict__ karr, const float* __restrict__ carry,
                float* __restrict__ out)   // out holds g on entry, h on exit
{
    const int gid = blockIdx.x * blockDim.x + threadIdx.x;   // < BH*CHUNKS
    const int c  = gid / BH;
    const int bh = gid % BH;
    const int b  = bh / H_;
    const int h  = bh % H_;
    size_t e = (size_t)b * S_ * H_ + (size_t)(c * CS) * H_ + h;

    float L = carry[c * BH + bh];
    #pragma unroll 4
    for (int s = 0; s < CS; ++s, e += H_) {
        const float kk = __builtin_nontemporal_load(&karr[e]);
        const float gg = out[e];
        float lcv, lvv;
        kg_transform(kk, gg, lcv, lvv);
        L = laef(lcv + L, lvv);
        __builtin_nontemporal_store(__expf(L), &out[e]);
    }
}

// ---------------------------------------------------------------------------
extern "C" void kernel_launch(void* const* d_in, const int* in_sizes, int n_in,
                              void* d_out, int out_size, void* d_ws, size_t ws_size,
                              hipStream_t stream)
{
    const float* x  = (const float*)d_in[0];
    const float* h0 = (const float*)d_in[1];
    const float* Wz = (const float*)d_in[2];
    const float* bz = (const float*)d_in[3];
    const float* Wh = (const float*)d_in[4];
    const float* bh = (const float*)d_in[5];
    float* out = (float*)d_out;                        // g during GEMM, h at end

    float*  karr  = (float*)d_ws;                      // (B,S,H) raw k
    float*  Asum  = karr + (size_t)MTOT * H_;
    float*  Lend  = Asum + (size_t)BH * CHUNKS;
    float*  carry = Lend + (size_t)BH * CHUNKS;
    __bf16* xbf   = (__bf16*)(carry + (size_t)BH * CHUNKS);   // MTOT*D_ bf16
    __bf16* wswz  = xbf + (size_t)MTOT * D_;                  // 2*WN bf16 (frag-major)

    convert_x_kernel<<<(MTOT * (D_ / 8)) / 256, 256, 0, stream>>>(x, xbf);
    swizzle_w_kernel<<<(2 * 64 * 32 * 32) / 256, 256, 0, stream>>>(Wz, Wh, wswz);

    dim3 gg(MTOT / BLK_M, H_ / BLK_N);                 // (256, 8)
    mingru_gemm_kernel<<<gg, 256, 0, stream>>>(xbf, wswz, bz, bh, karr, out);

    scan_pass1<<<(BH * CHUNKS) / 256, 256, 0, stream>>>(karr, out, Asum, Lend);
    scan_pass2<<<BH / 256, 256, 0, stream>>>(h0, Asum, Lend, carry);
    scan_pass3<<<(BH * CHUNKS) / 256, 256, 0, stream>>>(karr, carry, out);
}
